// MultiHeadCausalAttention_42923903156258
// MI455X (gfx1250) — matrix operand order repaired and checked by reading, not compiled, not run on hardware
//
#include <hip/hip_runtime.h>
#include <hip/hip_bf16.h>
#include <math.h>

// ---------------------------------------------------------------------------
// MultiHeadCausalAttention forward for MI455X (gfx1250), wave32 + WMMA f16.
// B=4, S=2048, D=1024, H=16, Hd=64.
// ---------------------------------------------------------------------------

typedef _Float16 h8  __attribute__((ext_vector_type(8)));
typedef _Float16 h16 __attribute__((ext_vector_type(16)));
typedef float    f8  __attribute__((ext_vector_type(8)));

#define NEG_INF (-__builtin_inff())

static __device__ __forceinline__ f8 fzero() {
    f8 z;
#pragma unroll
    for (int i = 0; i < 8; ++i) z[i] = 0.0f;
    return z;
}

// A fragment, 16x32 f16 (MxK), row-major source with leading dim ld.
// lane&15 = M row; lane>>4 selects K interleave half:
//   elems 0..7  -> K = (lane>>4)*8 + 0..7
//   elems 8..15 -> K = 16 + (lane>>4)*8 + 0..7
static __device__ __forceinline__ h16 frag_a(const _Float16* __restrict__ p, int ld) {
    const int lane = threadIdx.x & 31;
    const _Float16* row = p + (lane & 15) * ld + ((lane >> 4) << 3);
    h8 lo = *(const h8*)(row);
    h8 hi = *(const h8*)(row + 16);
    h16 f;
#pragma unroll
    for (int i = 0; i < 8; ++i) { f[i] = lo[i]; f[i + 8] = hi[i]; }
    return f;
}

// B fragment, 32x16 f16 (KxN). Source is "N-major": row n holds K contiguous
// (i.e. the transposed operand, e.g. torch-style (out,in) weights).
// lane&15 = N column; elems 0..7 -> K = (lane>>4)*8 + 0..7, elems 8..15 -> K = 16 + (lane>>4)*8 + 0..7.
static __device__ __forceinline__ h16 frag_b(const _Float16* __restrict__ p, int ld) {
    const int lane = threadIdx.x & 31;
    const _Float16* row = p + (lane & 15) * ld + ((lane >> 4) << 3);
    h8 lo = *(const h8*)(row);
    h8 hi = *(const h8*)(row + 16);
    h16 f;
#pragma unroll
    for (int i = 0; i < 8; ++i) { f[i] = lo[i]; f[i + 8] = hi[i]; }
    return f;
}

static __device__ __forceinline__ f8 wmma_f16(h16 a, h16 b, f8 c) {
    return __builtin_amdgcn_wmma_f32_16x16x32_f16(false, a, false, b, (short)0, c,
                                                  false, false);
}

// ---------------------------------------------------------------------------
// Kernel 1: f32 -> f16 downconvert (8 elems / thread / iter)
// ---------------------------------------------------------------------------
__global__ void cvt_f32_f16(const float* __restrict__ src,
                            _Float16* __restrict__ dst, int n8) {
    int i = blockIdx.x * blockDim.x + threadIdx.x;
    const int stride = gridDim.x * blockDim.x;
    for (; i < n8; i += stride) {
        const float4 a = ((const float4*)src)[2 * i + 0];
        const float4 b = ((const float4*)src)[2 * i + 1];
        h8 o;
        o[0] = (_Float16)a.x; o[1] = (_Float16)a.y;
        o[2] = (_Float16)a.z; o[3] = (_Float16)a.w;
        o[4] = (_Float16)b.x; o[5] = (_Float16)b.y;
        o[6] = (_Float16)b.z; o[7] = (_Float16)b.w;
        ((h8*)dst)[i] = o;
    }
}

// ---------------------------------------------------------------------------
// Shared GEMM mainloop: C(128x256) = A(128xK) * W^T(256xK), K=1024, K-step 32.
// Block = 256 thr (8 waves in a 2x4 grid), each wave owns a 64x64 register
// tile: 4 A-frags x 4 B-frags -> 16 v_wmma per K-step (1:1 wmma:ds_load_b128),
// 4x register reuse on both operands.
// ---------------------------------------------------------------------------
#define LDK 40  // f16 leading dim for LDS tiles (16B-aligned rows, skewed banks)

static __device__ __forceinline__ void gemm_mainloop_128x256(
    const _Float16* __restrict__ A,   // (M x 1024) row-major f16
    const _Float16* __restrict__ W,   // (N x 1024) row-major f16 (transposed op)
    int row0, int col0,
    _Float16* __restrict__ Ash,       // 128*LDK
    _Float16* __restrict__ Bsh,       // 256*LDK
    f8 (&acc)[4][4]) {
    const int tid = threadIdx.x, wid = tid >> 5;
    const int mw = (wid >> 2) * 64;   // wave row offset in tile (0 / 64)
    const int nw = (wid & 3) * 64;    // wave col offset in tile (0..192)

#pragma unroll
    for (int i = 0; i < 4; ++i)
#pragma unroll
        for (int j = 0; j < 4; ++j) acc[i][j] = fzero();

    const int arow = tid >> 1, ahalf = tid & 1;  // A: 128 rows x 2 halves of 16
    const _Float16* asrc0 = A + (row0 + arow) * 1024 + ahalf * 16;
    const _Float16* bsrc0 = W + (col0 + tid) * 1024;  // B: 1 row of 32 per thread

    for (int k0 = 0; k0 < 1024; k0 += 32) {
        // prefetch next K-step's sources into cache (global_prefetch_b8)
        if (k0 + 32 < 1024) {
            __builtin_prefetch(asrc0 + k0 + 32, 0, 1);
            __builtin_prefetch(bsrc0 + k0 + 32, 0, 1);
        }
        // stage A (2 x b128 / thread) and B (4 x b128 / thread) into LDS
        *(h8*)&Ash[arow * LDK + ahalf * 16 + 0] = *(const h8*)(asrc0 + k0 + 0);
        *(h8*)&Ash[arow * LDK + ahalf * 16 + 8] = *(const h8*)(asrc0 + k0 + 8);
#pragma unroll
        for (int q = 0; q < 4; ++q)
            *(h8*)&Bsh[tid * LDK + q * 8] = *(const h8*)(bsrc0 + k0 + q * 8);
        __syncthreads();

        h16 af[4];
#pragma unroll
        for (int i = 0; i < 4; ++i)
            af[i] = frag_a(&Ash[(mw + i * 16) * LDK], LDK);
#pragma unroll
        for (int j = 0; j < 4; ++j) {
            const h16 bf = frag_b(&Bsh[(nw + j * 16) * LDK], LDK);
#pragma unroll
            for (int i = 0; i < 4; ++i) acc[i][j] = wmma_f16(af[i], bf, acc[i][j]);
        }
        __syncthreads();
    }
}

// ---------------------------------------------------------------------------
// Kernel 2: QKV projection. M=8192(B*S), N=3072, K=1024.
// Output scattered: Q,K as (B,H,S,Hd) f16 (Q pre-scaled by 1/8), V transposed
// as (B,H,Hd,S) f16 so the PV B-frags later read contiguous keys.
// ---------------------------------------------------------------------------
__global__ __launch_bounds__(256) void qkv_gemm(
    const _Float16* __restrict__ xh, const _Float16* __restrict__ wh,
    const float* __restrict__ bqkv, _Float16* __restrict__ qh,
    _Float16* __restrict__ kh, _Float16* __restrict__ vt) {
    __shared__ __align__(16) _Float16 Ash[128 * LDK];
    __shared__ __align__(16) _Float16 Bsh[256 * LDK];

    const int tid = threadIdx.x, wid = tid >> 5, lane = tid & 31;
    const int row0 = blockIdx.x * 128;
    const int col0 = blockIdx.y * 256;

    f8 acc[4][4];
    gemm_mainloop_128x256(xh, wh, row0, col0, Ash, Bsh, acc);

    // Epilogue: each wave's 64-wide column strip is one section (Q/K/V), one head.
    const int cw0  = col0 + (wid & 3) * 64;
    const int sect = cw0 >> 10;          // 0=Q, 1=K, 2=V
    const int head = (cw0 & 1023) >> 6;  // 0..15
    const int mrow = row0 + (wid >> 2) * 64 + ((lane >> 4) << 3);
#pragma unroll
    for (int j = 0; j < 4; ++j) {
        const int d = j * 16 + (lane & 15);        // 0..63 within head
        const float bias = bqkv[cw0 + d];
#pragma unroll
        for (int i = 0; i < 4; ++i) {
#pragma unroll
            for (int r = 0; r < 8; ++r) {
                const int m = mrow + i * 16 + r;
                const int b = m >> 11, s = m & 2047;
                const float v = acc[i][j][r] + bias;
                if (sect == 0)
                    qh[(((b * 16 + head) * 2048 + s) << 6) + d] =
                        (_Float16)(v * 0.125f);
                else if (sect == 1)
                    kh[(((b * 16 + head) * 2048 + s) << 6) + d] = (_Float16)v;
                else
                    vt[(((b * 16 + head) * 64 + d) << 11) + s] = (_Float16)v;
            }
        }
    }
}

// ---------------------------------------------------------------------------
// Kernel 3: flash-style causal attention. 1 wave = 16 query rows x full Hd=64.
// Online softmax in f32; P transposed via per-wave LDS into an A-frag.
// ---------------------------------------------------------------------------
__global__ __launch_bounds__(128) void attn_kernel(
    const _Float16* __restrict__ qh, const _Float16* __restrict__ kh,
    const _Float16* __restrict__ vt, _Float16* __restrict__ oh) {
    __shared__ __align__(16) _Float16 Psh[4][16 * 32];

    const int tid = threadIdx.x, wid = tid >> 5, lane = tid & 31;
    const int b = blockIdx.z, h = blockIdx.y;
    const int s0 = (blockIdx.x * 4 + wid) * 16;

    const _Float16* qb = qh + (size_t)(b * 16 + h) * 2048 * 64;
    const _Float16* kb = kh + (size_t)(b * 16 + h) * 2048 * 64;
    const _Float16* vb = vt + (size_t)(b * 16 + h) * 64 * 2048;
    _Float16* psh = Psh[wid];

    const h16 qa0 = frag_a(qb + s0 * 64 + 0, 64);   // dims 0..31 (scale folded in)
    const h16 qa1 = frag_a(qb + s0 * 64 + 32, 64);  // dims 32..63

    f8 o[4];
#pragma unroll
    for (int c = 0; c < 4; ++c) o[c] = fzero();
    float mrun[8], lrun[8];
#pragma unroll
    for (int r = 0; r < 8; ++r) { mrun[r] = NEG_INF; lrun[r] = 0.0f; }

    const int key0 = lane & 15;
    const int rowb = s0 + ((lane >> 4) << 3);
    const int nkb = (s0 >> 5) + 1;  // 32-key blocks covering keys 0..s0+15

    for (int kblk = 0; kblk < nkb; ++kblk) {
        const int kbase = kblk * 32;
        // scores: two 16x16 tiles over 32 keys, K-dim = 64 head dims
        f8 sc0 = fzero(), sc1 = fzero();
        sc0 = wmma_f16(qa0, frag_b(kb + (kbase +  0) * 64 +  0, 64), sc0);
        sc0 = wmma_f16(qa1, frag_b(kb + (kbase +  0) * 64 + 32, 64), sc0);
        sc1 = wmma_f16(qa0, frag_b(kb + (kbase + 16) * 64 +  0, 64), sc1);
        sc1 = wmma_f16(qa1, frag_b(kb + (kbase + 16) * 64 + 32, 64), sc1);

        float alpha[8];
#pragma unroll
        for (int r = 0; r < 8; ++r) {
            const int srow = rowb + r;
            // causal mask
            if (kbase + key0 > srow)      sc0[r] = NEG_INF;
            if (kbase + 16 + key0 > srow) sc1[r] = NEG_INF;
            // row max across the 32 score columns (lanes 0..15 of each half)
            float v = fmaxf(sc0[r], sc1[r]);
            v = fmaxf(v, __shfl_xor(v, 1, 32));
            v = fmaxf(v, __shfl_xor(v, 2, 32));
            v = fmaxf(v, __shfl_xor(v, 4, 32));
            v = fmaxf(v, __shfl_xor(v, 8, 32));
            const float mn = fmaxf(mrun[r], v);
            alpha[r] = __expf(mrun[r] - mn);
            mrun[r] = mn;
            const float p0 = __expf(sc0[r] - mn);
            const float p1 = __expf(sc1[r] - mn);
            sc0[r] = p0; sc1[r] = p1;
            float sum = p0 + p1;
            sum += __shfl_xor(sum, 1, 32);
            sum += __shfl_xor(sum, 2, 32);
            sum += __shfl_xor(sum, 4, 32);
            sum += __shfl_xor(sum, 8, 32);
            lrun[r] = alpha[r] * lrun[r] + sum;
        }
#pragma unroll
        for (int c = 0; c < 4; ++c)
#pragma unroll
            for (int r = 0; r < 8; ++r) o[c][r] *= alpha[r];

        // transpose P (C layout -> A layout) through per-wave LDS
#pragma unroll
        for (int r = 0; r < 8; ++r) {
            const int prow = ((lane >> 4) << 3) + r;
            psh[prow * 32 + key0]      = (_Float16)sc0[r];
            psh[prow * 32 + 16 + key0] = (_Float16)sc1[r];
        }
        asm volatile("s_wait_dscnt 0" ::: "memory");
        const h16 pf = frag_a(psh, 32);

        // O += P(16x32) x V(32x64), V read transposed (Hd-major, keys contiguous)
#pragma unroll
        for (int c = 0; c < 4; ++c)
            o[c] = wmma_f16(pf, frag_b(vb + (c * 16) * 2048 + kbase, 2048), o[c]);
    }

    // normalize and write (B,S,H,Hd) = (B,S,D) f16
#pragma unroll
    for (int c = 0; c < 4; ++c) {
        const int d = h * 64 + c * 16 + (lane & 15);
#pragma unroll
        for (int r = 0; r < 8; ++r) {
            const int s = rowb + r;
            oh[((size_t)b * 2048 + s) * 1024 + d] = (_Float16)(o[c][r] / lrun[r]);
        }
    }
}

// ---------------------------------------------------------------------------
// Kernel 4: output projection. M=8192, N=1024, K=1024. f32 output + bias.
// ---------------------------------------------------------------------------
__global__ __launch_bounds__(256) void out_gemm(
    const _Float16* __restrict__ ah, const _Float16* __restrict__ wh,
    const float* __restrict__ bout, float* __restrict__ out) {
    __shared__ __align__(16) _Float16 Ash[128 * LDK];
    __shared__ __align__(16) _Float16 Bsh[256 * LDK];

    const int tid = threadIdx.x, wid = tid >> 5, lane = tid & 31;
    const int row0 = blockIdx.x * 128;
    const int col0 = blockIdx.y * 256;

    f8 acc[4][4];
    gemm_mainloop_128x256(ah, wh, row0, col0, Ash, Bsh, acc);

    const int cw0  = col0 + (wid & 3) * 64;
    const int mrow = row0 + (wid >> 2) * 64 + ((lane >> 4) << 3);
#pragma unroll
    for (int j = 0; j < 4; ++j) {
        const int col = cw0 + j * 16 + (lane & 15);
        const float bias = bout[col];
#pragma unroll
        for (int i = 0; i < 4; ++i) {
#pragma unroll
            for (int r = 0; r < 8; ++r)
                out[(size_t)(mrow + i * 16 + r) * 1024 + col] =
                    acc[i][j][r] + bias;
        }
    }
}

// ---------------------------------------------------------------------------
// Host-side launcher
// ---------------------------------------------------------------------------
extern "C" void kernel_launch(void* const* d_in, const int* in_sizes, int n_in,
                              void* d_out, int out_size, void* d_ws, size_t ws_size,
                              hipStream_t stream) {
    const float* x     = (const float*)d_in[0];   // (4,2048,1024)
    const float* wqkv  = (const float*)d_in[1];   // (3072,1024)
    const float* bqkv  = (const float*)d_in[2];   // (3072,)
    const float* wout  = (const float*)d_in[3];   // (1024,1024)
    const float* bout  = (const float*)d_in[4];   // (1024,)
    float* out         = (float*)d_out;           // (4,2048,1024)

    char* ws = (char*)d_ws;
    const size_t MB = 1024 * 1024;
    _Float16* xh    = (_Float16*)(ws + 0 * MB);   // 16 MB: x f16
    _Float16* wqkvh = (_Float16*)(ws + 16 * MB);  //  6 MB: w_qkv f16
    _Float16* wouth = (_Float16*)(ws + 22 * MB);  //  2 MB: w_out f16
    _Float16* qh    = (_Float16*)(ws + 24 * MB);  // 16 MB: Q (B,H,S,Hd), pre-scaled
    _Float16* kh    = (_Float16*)(ws + 40 * MB);  // 16 MB: K (B,H,S,Hd)
    _Float16* vt    = (_Float16*)(ws + 56 * MB);  // 16 MB: V^T (B,H,Hd,S)
    _Float16* oh    = (_Float16*)(ws + 72 * MB);  // 16 MB: attn out (B,S,D)

    cvt_f32_f16<<<2048, 256, 0, stream>>>(x,    xh,    (8192 * 1024) / 8);
    cvt_f32_f16<<<1024, 256, 0, stream>>>(wqkv, wqkvh, (3072 * 1024) / 8);
    cvt_f32_f16<<<512,  256, 0, stream>>>(wout, wouth, (1024 * 1024) / 8);

    qkv_gemm<<<dim3(64, 12), 256, 0, stream>>>(xh, wqkvh, bqkv, qh, kh, vt);
    attn_kernel<<<dim3(32, 16, 4), 128, 0, stream>>>(qh, kh, vt, oh);
    out_gemm<<<dim3(64, 4), 256, 0, stream>>>(oh, wouth, bout, out);
}
